// StandardHeteroSAGE_1099511628154
// MI455X (gfx1250) — matrix operand order repaired and checked by reading, not compiled, not run on hardware
//
#include <hip/hip_runtime.h>
#include <hip/hip_bf16.h>

typedef __attribute__((ext_vector_type(16))) _Float16 v16h;
typedef __attribute__((ext_vector_type(8)))  float    v8f;

#define N_USER 100000
#define N_ADS  50000
#define N_EDGES 800000
#define D_IN 256
#define H 128

// ---------------------------------------------------------------------------
// Pre-pack f32 row-major W[K,NOUT] into the WMMA f16 B-fragment order used here (contiguous k, not the ISA table order):
//   fragment (kt,nt): 32 lanes x 16 halves, contiguous per lane.
//   lane L: col n = nt*16 + (L&15); half j -> K = kt*32 + (L>>4)*16 + j
// One wave per (kt,nt) fragment.
// ---------------------------------------------------------------------------
template <int K, int NOUT>
__global__ __launch_bounds__(32) void pack_w_f16(const float* __restrict__ W,
                                                 _Float16* __restrict__ Wp) {
  constexpr int NT = NOUT / 16;
  const int lane = threadIdx.x;
  const int kt = blockIdx.x;              // 0..K/32-1
  const int nt = blockIdx.y;              // 0..NT-1
  const int n  = nt * 16 + (lane & 15);
  const int kb = kt * 32 + (lane >> 4) * 16;
  _Float16* __restrict__ dst = Wp + ((size_t)(kt * NT + nt) * 32 + lane) * 16;
#pragma unroll
  for (int j = 0; j < 16; ++j) dst[j] = (_Float16)W[(size_t)(kb + j) * NOUT + n];
}

// ---------------------------------------------------------------------------
// Row-strip GEMM: Y[N,NOUT] = relu?(X[N,K] @ W[K,NOUT] + bias)
// One wave computes a 16-row x NOUT-col strip with NT f32 accumulators:
// each A fragment loaded once (4x b128 + 8 cvt_pk) and reused across NT WMMAs;
// B fragments come pre-packed in f16 -> one contiguous v16h load each.
// A-fragment layout used here (contiguous k, matching packed B):
//   lane L: row m = L&15; half j -> K = k0 + (L>>4)*16 + j
// C/D layout: lane L: col n = L&15; VGPR r -> row m = r + 8*(L>>4)
// ---------------------------------------------------------------------------
template <int K, int NOUT>
__global__ __launch_bounds__(32) void gemm_rowstrip_wmma(
    const float* __restrict__ X, const _Float16* __restrict__ Wp,
    const float* __restrict__ bias, float* __restrict__ Y, int do_relu) {
  constexpr int NT = NOUT / 16;
  const int lane = threadIdx.x;          // 0..31
  const int row0 = blockIdx.x * 16;      // M tile
  const int m    = lane & 15;
  const int half = lane >> 4;            // 0 or 1

  const float* __restrict__ xrow = X + (size_t)(row0 + m) * K;

  v8f acc[NT];
#pragma unroll
  for (int nt = 0; nt < NT; ++nt) acc[nt] = (v8f){};

  for (int k0 = 0; k0 < K; k0 += 32) {
    if (k0 + 32 < K) __builtin_prefetch(xrow + k0 + 32, 0, 1);   // global_prefetch_b8

    // ---- A fragment: one contiguous 16-float run -> 4x b128 loads ----
    const int ka = k0 + half * 16;
    const float4 x0 = *(const float4*)(xrow + ka);
    const float4 x1 = *(const float4*)(xrow + ka + 4);
    const float4 x2 = *(const float4*)(xrow + ka + 8);
    const float4 x3 = *(const float4*)(xrow + ka + 12);
    v16h a;
    a[0]  = (_Float16)x0.x; a[1]  = (_Float16)x0.y; a[2]  = (_Float16)x0.z; a[3]  = (_Float16)x0.w;
    a[4]  = (_Float16)x1.x; a[5]  = (_Float16)x1.y; a[6]  = (_Float16)x1.z; a[7]  = (_Float16)x1.w;
    a[8]  = (_Float16)x2.x; a[9]  = (_Float16)x2.y; a[10] = (_Float16)x2.z; a[11] = (_Float16)x2.w;
    a[12] = (_Float16)x3.x; a[13] = (_Float16)x3.y; a[14] = (_Float16)x3.z; a[15] = (_Float16)x3.w;

    // ---- column tiles: pre-packed B fragments, A reused across all NT ----
    const _Float16* __restrict__ wpk =
        Wp + ((size_t)(k0 / 32) * NT * 32 + lane) * 16;
#pragma unroll
    for (int nt = 0; nt < NT; ++nt) {
      const v16h b = *(const v16h*)(wpk + (size_t)nt * 32 * 16);
      // (neg_a, A, neg_b, B, c_mod, C, reuse_a, reuse_b)
      acc[nt] = __builtin_amdgcn_wmma_f32_16x16x32_f16(
          false, a, false, b, (short)0, acc[nt], false, false);
    }
  }

#pragma unroll
  for (int nt = 0; nt < NT; ++nt) {
    const int ncol = nt * 16 + m;
    const float bv = bias[ncol];
#pragma unroll
    for (int r = 0; r < 8; ++r) {
      const int mo = r + 8 * half;
      float v = acc[nt][r] + bv;
      if (do_relu) v = fmaxf(v, 0.0f);
      Y[(size_t)(row0 + mo) * NOUT + ncol] = v;
    }
  }
}

// ---------------------------------------------------------------------------
__global__ void zero_f32(float* __restrict__ p, size_t n) {
  size_t i = (size_t)blockIdx.x * blockDim.x + threadIdx.x;
  size_t stride = (size_t)gridDim.x * blockDim.x;
  for (; i < n; i += stride) p[i] = 0.0f;
}

__global__ void count_edges(const int* __restrict__ dst, float* __restrict__ cnt, int nE) {
  int e = blockIdx.x * blockDim.x + threadIdx.x;
  if (e < nE) unsafeAtomicAdd(&cnt[dst[e]], 1.0f);
}

// One wave per edge; lane covers 4 consecutive columns (32*4 = H = 128).
__global__ void scatter_add_rows(const float* __restrict__ msg,
                                 const int* __restrict__ src,
                                 const int* __restrict__ dst,
                                 float* __restrict__ acc, int nE) {
  const int wave = (int)(((size_t)blockIdx.x * blockDim.x + threadIdx.x) >> 5);
  const int lane = threadIdx.x & 31;
  if (wave >= nE) return;
  const int s = src[wave];
  const int d = dst[wave];
  const float4 v = *(const float4*)(msg + (size_t)s * H + lane * 4);
  float* p = acc + (size_t)d * H + lane * 4;
  unsafeAtomicAdd(p + 0, v.x);
  unsafeAtomicAdd(p + 1, v.y);
  unsafeAtomicAdd(p + 2, v.z);
  unsafeAtomicAdd(p + 3, v.w);
}

// out = relu(LayerNorm(acc/max(cnt,1) + self)) ; one wave per row of 128.
__global__ void finalize_ln_relu(const float* __restrict__ acc,
                                 const float* __restrict__ cnt,
                                 const float* __restrict__ slf,
                                 const float* __restrict__ g,
                                 const float* __restrict__ b,
                                 float* __restrict__ out, int N) {
  const int row  = (int)(((size_t)blockIdx.x * blockDim.x + threadIdx.x) >> 5);
  const int lane = threadIdx.x & 31;
  if (row >= N) return;
  const float inv = 1.0f / fmaxf(cnt[row], 1.0f);
  const size_t base = (size_t)row * H + lane * 4;
  const float4 va = *(const float4*)(acc + base);
  const float4 vs = *(const float4*)(slf + base);
  float v0 = fmaf(va.x, inv, vs.x);
  float v1 = fmaf(va.y, inv, vs.y);
  float v2 = fmaf(va.z, inv, vs.z);
  float v3 = fmaf(va.w, inv, vs.w);

  float s = v0 + v1 + v2 + v3;
#pragma unroll
  for (int off = 16; off >= 1; off >>= 1) s += __shfl_xor(s, off, 32);
  const float mu = s * (1.0f / 128.0f);

  const float d0 = v0 - mu, d1 = v1 - mu, d2 = v2 - mu, d3 = v3 - mu;
  float q = d0 * d0 + d1 * d1 + d2 * d2 + d3 * d3;
#pragma unroll
  for (int off = 16; off >= 1; off >>= 1) q += __shfl_xor(q, off, 32);
  const float r = rsqrtf(q * (1.0f / 128.0f) + 1e-5f);

  const int c = lane * 4;
  float o0 = fmaxf(fmaf(d0 * r, g[c + 0], b[c + 0]), 0.0f);
  float o1 = fmaxf(fmaf(d1 * r, g[c + 1], b[c + 1]), 0.0f);
  float o2 = fmaxf(fmaf(d2 * r, g[c + 2], b[c + 2]), 0.0f);
  float o3 = fmaxf(fmaf(d3 * r, g[c + 3], b[c + 3]), 0.0f);
  *(float4*)(out + base) = make_float4(o0, o1, o2, o3);
}

// out[i] = dot(z[i,0:64], w2) + b2 ; one wave per row (2 elems per lane).
__global__ void head2_dot(const float* __restrict__ z, const float* __restrict__ w2,
                          const float* __restrict__ b2, float* __restrict__ out, int N) {
  const int row  = (int)(((size_t)blockIdx.x * blockDim.x + threadIdx.x) >> 5);
  const int lane = threadIdx.x & 31;
  if (row >= N) return;
  const float* zr = z + (size_t)row * 64;
  float s = zr[lane] * w2[lane] + zr[lane + 32] * w2[lane + 32];
#pragma unroll
  for (int off = 16; off >= 1; off >>= 1) s += __shfl_xor(s, off, 32);
  if (lane == 0) out[row] = s + b2[0];
}

// ---------------------------------------------------------------------------
extern "C" void kernel_launch(void* const* d_in, const int* in_sizes, int n_in,
                              void* d_out, int out_size, void* d_ws, size_t ws_size,
                              hipStream_t stream) {
  const float* xU    = (const float*)d_in[0];
  const float* xA    = (const float*)d_in[1];
  const int* eua_src = (const int*)d_in[2];
  const int* eua_dst = (const int*)d_in[3];
  const int* eau_src = (const int*)d_in[4];
  const int* eau_dst = (const int*)d_in[5];
  // params flattened in dict insertion order, leaves {w,b} / {g,b}
  const float* inU_w = (const float*)d_in[6];
  const float* inU_b = (const float*)d_in[7];
  const float* inA_w = (const float*)d_in[8];
  const float* inA_b = (const float*)d_in[9];
  // per layer L: base = 10 + 12*L:
  //  +0/+1 user_ads_neigh w/b   +2/+3 user_ads_self w/b
  //  +4/+5 ads_user_neigh w/b   +6/+7 ads_user_self w/b
  //  +8/+9 ln_User g/b          +10/+11 ln_AdsInfo g/b
  const float* h1_w = (const float*)d_in[34];
  const float* h1_b = (const float*)d_in[35];
  const float* h2_w = (const float*)d_in[36];
  const float* h2_b = (const float*)d_in[37];

  // ---- workspace layout ----
  float* ws = (float*)d_ws;
  float* hU   = ws;                         ws += (size_t)N_USER * H;   // current user feats
  float* hA   = ws;                         ws += (size_t)N_ADS  * H;   // current ads feats
  float* bufU = ws;                         ws += (size_t)N_USER * H;   // msgU then selfU
  float* bufA = ws;                         ws += (size_t)N_ADS  * H;   // msgA then selfA
  float* accU = ws;                         ws += (size_t)N_USER * H;
  float* accA = ws;                         ws += (size_t)N_ADS  * H;
  float* cntU = ws;                         ws += N_USER;
  float* cntA = ws;                         ws += N_ADS;
  float* zH   = ws;                         ws += (size_t)N_ADS * 64;
  // packed f16 weight fragments
  _Float16* wp = (_Float16*)ws;
  _Float16* wp_inU = wp;                    wp += (size_t)D_IN * H;     // 256x128
  _Float16* wp_inA = wp;                    wp += (size_t)D_IN * H;
  _Float16* wp_L[2][4];                                                // nUA,sUA,nAU,sAU
  for (int L = 0; L < 2; ++L)
    for (int i = 0; i < 4; ++i) { wp_L[L][i] = wp; wp += (size_t)H * H; }
  _Float16* wp_h1 = wp;                     wp += (size_t)H * 64;

  const dim3 wave(32, 1, 1);
  const int ZB = 1024;

  // ---- pack all weights into WMMA B-fragment layout (tiny, once per call) ----
  pack_w_f16<D_IN, H><<<dim3(D_IN / 32, H / 16), wave, 0, stream>>>(inU_w, wp_inU);
  pack_w_f16<D_IN, H><<<dim3(D_IN / 32, H / 16), wave, 0, stream>>>(inA_w, wp_inA);
  for (int L = 0; L < 2; ++L) {
    const int base = 10 + 12 * L;
    pack_w_f16<H, H><<<dim3(H / 32, H / 16), wave, 0, stream>>>((const float*)d_in[base + 0], wp_L[L][0]); // nUA
    pack_w_f16<H, H><<<dim3(H / 32, H / 16), wave, 0, stream>>>((const float*)d_in[base + 2], wp_L[L][1]); // sUA
    pack_w_f16<H, H><<<dim3(H / 32, H / 16), wave, 0, stream>>>((const float*)d_in[base + 4], wp_L[L][2]); // nAU
    pack_w_f16<H, H><<<dim3(H / 32, H / 16), wave, 0, stream>>>((const float*)d_in[base + 6], wp_L[L][3]); // sAU
  }
  pack_w_f16<H, 64><<<dim3(H / 32, 64 / 16), wave, 0, stream>>>(h1_w, wp_h1);

  // ---- input projections + relu (256 -> 128) ----
  gemm_rowstrip_wmma<D_IN, H><<<N_USER / 16, wave, 0, stream>>>(xU, wp_inU, inU_b, hU, 1);
  gemm_rowstrip_wmma<D_IN, H><<<N_ADS  / 16, wave, 0, stream>>>(xA, wp_inA, inA_b, hA, 1);

  for (int L = 0; L < 2; ++L) {
    const int base = 10 + 12 * L;
    const float* nUA_b = (const float*)d_in[base + 1];
    const float* sUA_b = (const float*)d_in[base + 3];
    const float* nAU_b = (const float*)d_in[base + 5];
    const float* sAU_b = (const float*)d_in[base + 7];
    const float* lnU_g = (const float*)d_in[base + 8];
    const float* lnU_b = (const float*)d_in[base + 9];
    const float* lnA_g = (const float*)d_in[base + 10];
    const float* lnA_b = (const float*)d_in[base + 11];

    // user->ads messages: project then scatter-mean into accA
    gemm_rowstrip_wmma<H, H><<<N_USER / 16, wave, 0, stream>>>(hU, wp_L[L][0], nUA_b, bufU, 0);
    zero_f32<<<256, ZB, 0, stream>>>(accA, (size_t)N_ADS * H);
    zero_f32<<<64,  ZB, 0, stream>>>(cntA, (size_t)N_ADS);
    count_edges<<<(N_EDGES + 255) / 256, 256, 0, stream>>>(eua_dst, cntA, N_EDGES);
    scatter_add_rows<<<(N_EDGES + 7) / 8, 256, 0, stream>>>(bufU, eua_src, eua_dst, accA, N_EDGES);

    // ads->user messages
    gemm_rowstrip_wmma<H, H><<<N_ADS / 16, wave, 0, stream>>>(hA, wp_L[L][2], nAU_b, bufA, 0);
    zero_f32<<<256, ZB, 0, stream>>>(accU, (size_t)N_USER * H);
    zero_f32<<<128, ZB, 0, stream>>>(cntU, (size_t)N_USER);
    count_edges<<<(N_EDGES + 255) / 256, 256, 0, stream>>>(eau_dst, cntU, N_EDGES);
    scatter_add_rows<<<(N_EDGES + 7) / 8, 256, 0, stream>>>(bufA, eau_src, eau_dst, accU, N_EDGES);

    // self transforms (msg buffers free once their scatter is issued in-order)
    gemm_rowstrip_wmma<H, H><<<N_ADS  / 16, wave, 0, stream>>>(hA, wp_L[L][1], sUA_b, bufA, 0);
    gemm_rowstrip_wmma<H, H><<<N_USER / 16, wave, 0, stream>>>(hU, wp_L[L][3], sAU_b, bufU, 0);

    // mean + self + layernorm + relu -> new features (overwrites hU/hA)
    finalize_ln_relu<<<N_USER / 8, 256, 0, stream>>>(accU, cntU, bufU, lnU_g, lnU_b, hU, N_USER);
    finalize_ln_relu<<<N_ADS  / 8, 256, 0, stream>>>(accA, cntA, bufA, lnA_g, lnA_b, hA, N_ADS);
  }

  // head: relu(128->64) then 64->1
  gemm_rowstrip_wmma<H, 64><<<N_ADS / 16, wave, 0, stream>>>(hA, wp_h1, h1_b, zH, 1);
  head2_dot<<<N_ADS / 8, 256, 0, stream>>>(zH, h2_w, h2_b, (float*)d_out, N_ADS);
}
